// ATSBlock_15015205667055
// MI455X (gfx1250) — matrix operand order repaired and checked by reading, not compiled, not run on hardware
//
#include <hip/hip_runtime.h>
#include <math.h>

// ---------------------------------------------------------------------------
// Types for CDNA5 WMMA (wave32)
// ---------------------------------------------------------------------------
typedef __bf16 bf16_t;
typedef bf16_t v16bf __attribute__((ext_vector_type(16)));
typedef float  v8f   __attribute__((ext_vector_type(8)));
typedef unsigned int u32x4 __attribute__((ext_vector_type(4)));

union V16 {
    v16bf v;
    unsigned short u[16];
    u32x4 q[2];
};

__device__ __forceinline__ unsigned short f2bf(float f) {
    unsigned int u = __float_as_uint(f);
    unsigned int r = u + 0x7fffu + ((u >> 16) & 1u);   // RNE
    return (unsigned short)(r >> 16);
}

// Tensor Data Mover availability (device pass only; host pass -> fallback)
#if defined(__AMDGCN__) && __has_builtin(__builtin_amdgcn_tensor_load_to_lds) && \
    __has_builtin(__builtin_amdgcn_s_wait_tensorcnt)
#define HAVE_TDM 1
#else
#define HAVE_TDM 0
#endif

#if HAVE_TDM
typedef unsigned int tdm_u32x4 __attribute__((ext_vector_type(4)));
typedef int          tdm_i32x8 __attribute__((ext_vector_type(8)));
typedef int          tdm_i32x4 __attribute__((ext_vector_type(4)));

// Issue one TDM descriptor: 2-D tile, 16 elems (32B) x 32 rows of bf16,
// global row stride = ldb elements, into LDS row-major at lds_off.
// D# layout per CDNA5 ISA ch.8: group0 = {count/type, lds_addr, global_addr},
// group1 = {data_size, tensor dims, tile dims, dim0 stride}.
__device__ __forceinline__ void tdm_load_panel(const unsigned short* gsrc,
                                               unsigned int lds_off, int ldb) {
    unsigned long long ga = (unsigned long long)gsrc;
    tdm_u32x4 g0;
    g0[0] = 1u;                                         // count=1, user mode
    g0[1] = lds_off;                                    // lds_addr
    g0[2] = (unsigned int)ga;                           // global_addr[31:0]
    g0[3] = (unsigned int)((ga >> 32) & 0x01ffffffull)  // global_addr[56:32]
            | (2u << 30);                               // type = 2 ("image")
    tdm_i32x8 g1;
    g1[0] = 1 << 16;            // workgroup_mask=0, data_size=1 (2 bytes)
    g1[1] = 16 << 16;           // tensor_dim0 = 16 (low 16 bits)
    g1[2] = 32 << 16;           // tensor_dim1 = 32 (low 16 bits)
    g1[3] = 16 << 16;           // tile_dim0 = 16
    g1[4] = 32;                 // tile_dim1 = 32, tile_dim2 = 0
    g1[5] = ldb;                // tensor_dim0_stride[31:0]
    g1[6] = 0;                  // stride high bits, dim1 stride = 0
    g1[7] = 0;
    tdm_i32x4 g2 = {0, 0, 0, 0};
    tdm_i32x4 g3 = {0, 0, 0, 0};
#if __clang_major__ >= 23
    tdm_i32x8 g4 = {0, 0, 0, 0, 0, 0, 0, 0};
    __builtin_amdgcn_tensor_load_to_lds(g0, g1, g2, g3, g4, 0);
#else
    __builtin_amdgcn_tensor_load_to_lds(g0, g1, g2, g3, 0);
#endif
}
#endif  // HAVE_TDM

// Problem constants
static constexpr int   Bz  = 64;
static constexpr int   Nn  = 197;
static constexpr int   Cc  = 768;
static constexpr int   Hh  = 12;
static constexpr int   HD  = 64;
static constexpr int   C3  = 2304;
static constexpr int   C4  = 3072;
static constexpr size_t BN = (size_t)Bz * Nn;          // 12608 (788 * 16, exact)

// ---------------------------------------------------------------------------
// Workspace layout (bytes).  Big buffers reused across phases.
// ---------------------------------------------------------------------------
static constexpr size_t o_xnbf  = 0;                                     // LN1 out bf16, later xo bf16
static constexpr size_t o_wqkv  = o_xnbf  + BN * Cc * 2;
static constexpr size_t o_wproj = o_wqkv  + (size_t)Cc * C3 * 2;
static constexpr size_t o_wfc1  = o_wproj + (size_t)Cc * Cc * 2;
static constexpr size_t o_wfc2  = o_wfc1  + (size_t)Cc * C4 * 2;
static constexpr size_t o_qkv   = o_wfc2  + (size_t)C4 * Cc * 2;         // f32 qkv, later GELU bf16
static constexpr size_t o_attn  = o_qkv   + BN * C3 * 4;                 // f32 attn, later xo f32
static constexpr size_t o_attns = o_attn  + (size_t)Bz * Hh * Nn * Nn * 4; // bf16 attn_s (197 x 224 pad), later LN2 out
static constexpr size_t o_selx  = o_attns + (size_t)Bz * Hh * Nn * 224 * 2;
static constexpr size_t o_x2    = o_selx  + BN * Cc * 4;
static constexpr size_t o_sraw  = o_x2    + BN * Cc * 4;
static constexpr size_t o_src   = o_sraw  + 65536;
static constexpr size_t o_pol   = o_src   + 65536;

// ---------------------------------------------------------------------------
// f32 -> bf16 conversion (weights, activations)
// ---------------------------------------------------------------------------
__global__ void f32_to_bf16_kernel(const float* __restrict__ in,
                                   unsigned short* __restrict__ out, size_t n) {
    for (size_t i = (size_t)blockIdx.x * blockDim.x + threadIdx.x; i < n;
         i += (size_t)gridDim.x * blockDim.x)
        out[i] = f2bf(in[i]);
}

// ---------------------------------------------------------------------------
// LayerNorm (one block per row of 768) -> bf16
// ---------------------------------------------------------------------------
__global__ __launch_bounds__(256) void layernorm_to_bf16(
        const float* __restrict__ x, const float* __restrict__ g,
        const float* __restrict__ be, unsigned short* __restrict__ out) {
    const int row = blockIdx.x;
    const int t   = threadIdx.x;
    __shared__ float r1[256], r2[256], mv[2];
    const float* xr = x + (size_t)row * Cc;
    float s = 0.f, s2 = 0.f;
    for (int c = t; c < Cc; c += 256) { float v = xr[c]; s += v; s2 += v * v; }
    r1[t] = s; r2[t] = s2;
    for (int off = 128; off > 0; off >>= 1) {
        __syncthreads();
        if (t < off) { r1[t] += r1[t + off]; r2[t] += r2[t + off]; }
    }
    __syncthreads();
    if (t == 0) {
        float m   = r1[0] / (float)Cc;
        float var = r2[0] / (float)Cc - m * m;
        mv[0] = m; mv[1] = rsqrtf(var + 1e-5f);
    }
    __syncthreads();
    const float m = mv[0], rs = mv[1];
    unsigned short* orow = out + (size_t)row * Cc;
    for (int c = t; c < Cc; c += 256)
        orow[c] = f2bf((xr[c] - m) * rs * g[c] + be[c]);
}

// ---------------------------------------------------------------------------
// Generic bf16 WMMA GEMM.  Block = 8 waves = 128 M-rows sharing one 16-col
// N tile; the shared 32x16 bf16 B panel is double buffered in LDS.
//   TDM path: wave 0 issues one tensor_load_to_lds per K-step (DMA engine
//             fills LDS while waves run WMMA), drains TENSORcnt, barrier.
//   Fallback: cooperative coalesced b32 global loads + transposed LDS store,
//             each lane then reads its fragment as two ds_load_b128.
// MODE 0: out = acc * rowScale                    (qkv, policy scale)
// MODE 1: out = (acc+bias)*rowScale + resid       (proj -> x2)
// MODE 2: out = gelu(acc+bias) -> bf16            (fc1)
// MODE 3: out = (acc+bias+resid)*rowScale         (fc2 -> final out)
// ---------------------------------------------------------------------------
template <int MODE>
__global__ __launch_bounds__(256) void gemm_bf16_wmma(
        const unsigned short* __restrict__ A, int lda,
        const unsigned short* __restrict__ Bm, int ldb,
        const float* __restrict__ bias,
        const float* __restrict__ rowScale,
        const float* __restrict__ resid,
        float* __restrict__ Cf,
        unsigned short* __restrict__ Cbf,
        int Mrows, int Kdim, int ldc) {
#if HAVE_TDM
    __shared__ unsigned short BtR[2][32][16];  // [buf][k][n] row-major (TDM fill)
#else
    __shared__ unsigned short Bt[2][16][48];   // [buf][n][k] transposed, padded
#endif

    const int t     = threadIdx.x;
    const int lane  = t & 31;
    const int wave  = t >> 5;
    const int mtile = blockIdx.y * 8 + wave;
    const bool mvalid = (mtile * 16) < Mrows;
    const int mt    = mvalid ? mtile : (Mrows / 16 - 1);   // clamp, keep barriers
    const int l15   = lane & 15;
    const int hi    = lane >> 4;
    const int col   = blockIdx.x * 16 + l15;
    const int n0    = blockIdx.x * 16;
    const unsigned short* Arow = A + (size_t)(mt * 16 + l15) * lda;

#if HAVE_TDM
    if (wave == 0) {
        tdm_load_panel(Bm + n0, (unsigned int)(unsigned long long)&BtR[0][0][0], ldb);
        __builtin_amdgcn_s_wait_tensorcnt(0);
    }
#else
    // staging map: 512 panel elems / 256 threads = 2 consecutive n per thread
    const int sk = t >> 3;           // k row in panel 0..31
    const int sn = (t & 7) * 2;      // n col 0,2,...,14
    {   // prologue: stage panel kb=0 into buffer 0 (one coalesced b32 each)
        unsigned int w = *(const unsigned int*)(Bm + (size_t)sk * ldb + n0 + sn);
        Bt[0][sn][sk]     = (unsigned short)(w & 0xffffu);
        Bt[0][sn + 1][sk] = (unsigned short)(w >> 16);
    }
#endif
    __syncthreads();

    v8f acc = {0.f, 0.f, 0.f, 0.f, 0.f, 0.f, 0.f, 0.f};
    int buf = 0;
    for (int kb = 0; kb < Kdim; kb += 32) {
        // stage next panel into the other buffer (overlaps WMMA below)
        const bool more = (kb + 32) < Kdim;
#if HAVE_TDM
        if (more && wave == 0)
            tdm_load_panel(Bm + (size_t)(kb + 32) * ldb + n0,
                           (unsigned int)(unsigned long long)&BtR[buf ^ 1][0][0],
                           ldb);
#else
        if (more) {
            unsigned int w = *(const unsigned int*)(
                    Bm + (size_t)(kb + 32 + sk) * ldb + n0 + sn);
            Bt[buf ^ 1][sn][sk]     = (unsigned short)(w & 0xffffu);
            Bt[buf ^ 1][sn + 1][sk] = (unsigned short)(w >> 16);
        }
#endif
        V16 a, bm;
        const unsigned short* ap = Arow + kb + hi * 16;
        __builtin_prefetch(ap + 32, 0, 0);
        a.q[0] = *(const u32x4*)(ap);
        a.q[1] = *(const u32x4*)(ap + 8);
#if HAVE_TDM
#pragma unroll
        for (int tt = 0; tt < 16; ++tt) bm.u[tt] = BtR[buf][hi * 16 + tt][l15];
#else
        const unsigned short* bp = &Bt[buf][l15][hi * 16];
        bm.q[0] = *(const u32x4*)(bp);        // ds_load_b128
        bm.q[1] = *(const u32x4*)(bp + 8);    // ds_load_b128
#endif
        acc = __builtin_amdgcn_wmma_f32_16x16x32_bf16(
                false, a.v, false, bm.v, (short)0, acc, false, false);
#if HAVE_TDM
        if (more && wave == 0) __builtin_amdgcn_s_wait_tensorcnt(0);
#endif
        __syncthreads();
        buf ^= 1;
    }

    if (!mvalid) return;
#pragma unroll
    for (int r = 0; r < 8; ++r) {
        const int orow = mtile * 16 + r + hi * 8;
        const size_t oidx = (size_t)orow * ldc + col;
        float v = acc[r];
        if (MODE == 0) {
            Cf[oidx] = v * rowScale[orow];
        } else if (MODE == 1) {
            Cf[oidx] = (v + bias[col]) * rowScale[orow] + resid[oidx];
        } else if (MODE == 2) {
            v += bias[col];
            v = 0.5f * v * (1.0f + erff(v * 0.70710678118654752f));
            Cbf[oidx] = f2bf(v);
        } else {
            Cf[oidx] = (v + bias[col] + resid[oidx]) * rowScale[orow];
        }
    }
}

// ---------------------------------------------------------------------------
// attn_raw = scale * q @ k^T, per (b,h), bf16 WMMA with on-the-fly conversion
// grid (13 ntiles, 2, B*H), block 256
// ---------------------------------------------------------------------------
__global__ __launch_bounds__(256) void attn_qk_wmma(
        const float* __restrict__ qkv, float* __restrict__ attn) {
    const int lane  = threadIdx.x & 31;
    const int wave  = threadIdx.x >> 5;
    const int mtile = blockIdx.y * 8 + wave;
    if (mtile >= 13) return;
    const int bh = blockIdx.z, b = bh / Hh, h = bh % Hh;
    const int l15 = lane & 15, hi = lane >> 4;
    const int i = mtile * 16 + l15;
    const int j = blockIdx.x * 16 + l15;
    const int ic = i > Nn - 1 ? Nn - 1 : i;
    const int jc = j > Nn - 1 ? Nn - 1 : j;
    const float* qr = qkv + (size_t)(b * Nn + ic) * C3 + h * HD;
    const float* kr = qkv + (size_t)(b * Nn + jc) * C3 + Cc + h * HD;

    v8f acc = {0.f, 0.f, 0.f, 0.f, 0.f, 0.f, 0.f, 0.f};
#pragma unroll
    for (int kb = 0; kb < HD; kb += 32) {
        V16 a, bm;
        const float* ap = qr + kb + hi * 16;
#pragma unroll
        for (int tt = 0; tt < 8; ++tt) a.u[tt]     = f2bf(ap[tt]);
#pragma unroll
        for (int tt = 0; tt < 8; ++tt) a.u[8 + tt] = f2bf(ap[8 + tt]);
        const float* bp = kr + kb + hi * 16;
#pragma unroll
        for (int tt = 0; tt < 16; ++tt) bm.u[tt] = f2bf(bp[tt]);
        acc = __builtin_amdgcn_wmma_f32_16x16x32_bf16(
                false, a.v, false, bm.v, (short)0, acc, false, false);
    }
#pragma unroll
    for (int r = 0; r < 8; ++r) {
        const int iout = mtile * 16 + r + hi * 8;
        if (iout < Nn && j < Nn)
            attn[((size_t)bh * Nn + iout) * Nn + j] = acc[r] * 0.125f;
    }
}

// ---------------------------------------------------------------------------
// softmax_with_policy, one block per attention row
// ---------------------------------------------------------------------------
__global__ __launch_bounds__(256) void softmax_policy_kernel(
        float* __restrict__ attn, const float* __restrict__ policy) {
    const int gid = blockIdx.x;
    const int i = gid % Nn, bh = gid / Nn, b = bh / Hh;
    float* row = attn + ((size_t)bh * Nn + i) * Nn;
    const int t = threadIdx.x;
    __shared__ float red[256];
    float v = (t < Nn) ? row[t] : -3.4e38f;
    red[t] = v;
    for (int off = 128; off > 0; off >>= 1) {
        __syncthreads();
        if (t < off) red[t] = fmaxf(red[t], red[t + off]);
    }
    __syncthreads();
    const float mx = red[0];
    __syncthreads();
    float a = 0.f;
    if (t < Nn) {
        float p  = policy[b * Nn + t];
        float ap = p + (1.f - p) * ((i == t) ? 1.f : 0.f);
        a = expf(v - mx) * ap;
    }
    red[t] = a;
    for (int off = 128; off > 0; off >>= 1) {
        __syncthreads();
        if (t < off) red[t] += red[t + off];
    }
    __syncthreads();
    const float s = red[0];
    if (t < Nn) row[t] = (a + 1e-6f / (float)Nn) / (s + 1e-6f);
}

// ---------------------------------------------------------------------------
// score = (sum_h cls_attn[h,n]) * ||v[b,n,:]||  (one block per (b,n))
// ---------------------------------------------------------------------------
__global__ __launch_bounds__(256) void score_kernel(
        const float* __restrict__ qkv, const float* __restrict__ attn,
        float* __restrict__ sraw) {
    const int n = blockIdx.x % Nn, b = blockIdx.x / Nn;
    __shared__ float red[256];
    const float* vrow = qkv + (size_t)(b * Nn + n) * C3 + 2 * Cc;
    float s = 0.f;
    for (int c = threadIdx.x; c < Cc; c += 256) { float v = vrow[c]; s += v * v; }
    red[threadIdx.x] = s;
    for (int off = 128; off > 0; off >>= 1) {
        __syncthreads();
        if (threadIdx.x < off) red[threadIdx.x] += red[threadIdx.x + off];
    }
    __syncthreads();
    if (threadIdx.x == 0) {
        float vn = sqrtf(red[0]);
        float cls = 0.f;
        for (int h = 0; h < Hh; ++h)
            cls += attn[((size_t)(b * Hh + h) * Nn + 0) * Nn + n];
        sraw[b * Nn + n] = cls * vn;
    }
}

// ---------------------------------------------------------------------------
// Bitonic sort of 256 (key asc, carries val)
// ---------------------------------------------------------------------------
__device__ void bitonic256(float* key, int* val, int t) {
    for (int k = 2; k <= 256; k <<= 1)
        for (int j = k >> 1; j > 0; j >>= 1) {
            __syncthreads();
            int ixj = t ^ j;
            if (ixj > t) {
                bool up = ((t & k) == 0);
                float a = key[t], bK = key[ixj];
                if ((a > bK) == up) {
                    key[t] = bK; key[ixj] = a;
                    int tv = val[t]; val[t] = val[ixj]; val[ixj] = tv;
                }
            }
        }
    __syncthreads();
}

// ---------------------------------------------------------------------------
// Full ATS sampling pipeline, one block per batch: normalize -> argsort ->
// cumsum -> ncdf -> ys -> argmin match -> unique -> src rows + policy
// ---------------------------------------------------------------------------
__global__ __launch_bounds__(256) void ats_sample_kernel(
        const float* __restrict__ sraw, int* __restrict__ srcrow,
        float* __restrict__ pol) {
    const int b = blockIdx.x, t = threadIdx.x;
    const int M = Nn - 1;  // 196
    __shared__ float key[256];
    __shared__ int   val[256];
    __shared__ float ncdf[256];
    __shared__ int   ord[256];
    __shared__ float sc[2];

    float sv = (t < M) ? sraw[b * Nn + 1 + t] : 0.f;
    key[t] = sv;
    for (int off = 128; off > 0; off >>= 1) {
        __syncthreads();
        if (t < off) key[t] += key[t + off];
    }
    __syncthreads();
    const float ssum = key[0];
    __syncthreads();

    key[t] = (t < M) ? sv / ssum : 3.0e38f;
    val[t] = t;
    __syncthreads();
    bitonic256(key, val, t);          // ascending: sorted scores + order
    ord[t] = val[t];
    __syncthreads();

    if (t == 0) {
        float c = 0.f;
        for (int i = 0; i < M; ++i) { c += key[i]; ncdf[i] = c; }
        const float c0 = ncdf[0], inv = 1.0f / (ncdf[M - 1] - c0);
        float ystart = 3.0e38f;
        for (int i = 0; i < M; ++i) {
            float nv = (ncdf[i] - c0) * inv;
            ncdf[i] = nv;
            float cand = nv + ((nv == 0.f) ? 1e8f : 0.f);
            ystart = fminf(ystart, cand);
        }
        sc[0] = ystart;
    }
    __syncthreads();
    const float ystart = sc[0];

    int pick = 0;
    if (t < M) {
        float ys = ystart + (float)t * (1.0f - ystart) / (float)(Nn - 2);
        float bd = 3.0e38f;
        for (int k = 0; k < M; ++k) {
            float d = fabsf(ys - ncdf[k]);
            if (d < bd) { bd = d; pick = k; }
        }
    }
    __syncthreads();
    key[t] = (t < M) ? (float)pick : 3.0e38f;
    val[t] = 0;
    __syncthreads();
    bitonic256(key, val, t);          // sorted tokens_to_pick

    int cur = 0, nxt = 0;
    if (t < M) {
        cur = (int)key[t];
        nxt = (t == M - 1) ? 1 : (int)key[t + 1];
    }
    __syncthreads();
    key[t] = (t < M) ? (float)((nxt == cur) ? M : cur) : 3.0e38f;
    __syncthreads();
    bitonic256(key, val, t);          // uniq, ascending

    if (t < M) {
        int u  = (int)key[t];
        int sr = (u >= M) ? -1 : (1 + ord[u]);
        srcrow[b * M + t]   = sr;
        pol[b * Nn + 1 + t] = (sr >= 0) ? 1.0f : 0.0f;
    }
    if (t == 0) pol[b * Nn] = 1.0f;
}

// ---------------------------------------------------------------------------
// Gather attn rows -> attn_s bf16 [bh][197 rows][224 cols, zero padded]
// ---------------------------------------------------------------------------
__global__ __launch_bounds__(256) void gather_attn_kernel(
        const float* __restrict__ attn, const int* __restrict__ srcrow,
        unsigned short* __restrict__ attns) {
    const int jout = blockIdx.x;           // 0..196
    const int bh = blockIdx.y, b = bh / Hh;
    const int src = (jout == 0) ? 0 : srcrow[b * (Nn - 1) + (jout - 1)];
    const float* srow = attn + ((size_t)bh * Nn + (src < 0 ? 0 : src)) * Nn;
    unsigned short* orow = attns + ((size_t)bh * Nn + jout) * 224;
    for (int c = threadIdx.x; c < 224; c += 256) {
        float v = (c < Nn && (src >= 0 || jout == 0)) ? srow[c] : 0.f;
        orow[c] = f2bf(v);
    }
}

// ---------------------------------------------------------------------------
// Gather x rows -> selected_x f32 [b][197][768]
// ---------------------------------------------------------------------------
__global__ __launch_bounds__(256) void gather_x_kernel(
        const float* __restrict__ x, const int* __restrict__ srcrow,
        float* __restrict__ selx) {
    const int jout = blockIdx.x, b = blockIdx.y;
    const int src = (jout == 0) ? 0 : srcrow[b * (Nn - 1) + (jout - 1)];
    const float* srow = x + ((size_t)b * Nn + (src < 0 ? 0 : src)) * Cc;
    float* orow = selx + ((size_t)b * Nn + jout) * Cc;
    for (int c = threadIdx.x; c < Cc; c += 256)
        orow[c] = (jout == 0 || src >= 0) ? srow[c] : 0.f;
}

// ---------------------------------------------------------------------------
// xo = attn_s @ v, per (b,h).  A = attn_s bf16 (lda=224, K-padded zeros),
// B = v from qkv f32 with on-the-fly bf16 conversion.  grid (4, 2, B*H)
// ---------------------------------------------------------------------------
__global__ __launch_bounds__(256) void attn_av_wmma(
        const unsigned short* __restrict__ attns, const float* __restrict__ qkv,
        float* __restrict__ xo) {
    const int lane  = threadIdx.x & 31;
    const int wave  = threadIdx.x >> 5;
    const int mtile = blockIdx.y * 8 + wave;
    if (mtile >= 13) return;
    const int bh = blockIdx.z, b = bh / Hh, h = bh % Hh;
    const int l15 = lane & 15, hi = lane >> 4;
    const int d = blockIdx.x * 16 + l15;
    const int i = mtile * 16 + l15;
    const int ic = i > Nn - 1 ? Nn - 1 : i;
    const unsigned short* Ar = attns + ((size_t)bh * Nn + ic) * 224;

    v8f acc = {0.f, 0.f, 0.f, 0.f, 0.f, 0.f, 0.f, 0.f};
    for (int kb = 0; kb < 224; kb += 32) {
        V16 a, bm;
        const unsigned short* ap = Ar + kb + hi * 16;
        a.q[0] = *(const u32x4*)(ap);
        a.q[1] = *(const u32x4*)(ap + 8);
#pragma unroll
        for (int tt = 0; tt < 16; ++tt) {
            int m = kb + hi * 16 + tt;
            if (m > Nn - 1) m = Nn - 1;   // A is zero there; value irrelevant
            bm.u[tt] = f2bf(qkv[(size_t)(b * Nn + m) * C3 + 2 * Cc + h * HD + d]);
        }
        acc = __builtin_amdgcn_wmma_f32_16x16x32_bf16(
                false, a.v, false, bm.v, (short)0, acc, false, false);
    }
#pragma unroll
    for (int r = 0; r < 8; ++r) {
        const int iout = mtile * 16 + r + hi * 8;
        if (iout < Nn)
            xo[((size_t)(b * Nn + iout)) * Cc + h * HD + d] = acc[r];
    }
}

// ---------------------------------------------------------------------------
// Host-side launch sequence
// ---------------------------------------------------------------------------
extern "C" void kernel_launch(void* const* d_in, const int* in_sizes, int n_in,
                              void* d_out, int out_size, void* d_ws, size_t ws_size,
                              hipStream_t stream) {
    const float* x      = (const float*)d_in[0];
    const float* policy = (const float*)d_in[1];
    const float* w_qkv  = (const float*)d_in[2];
    const float* w_proj = (const float*)d_in[3];
    const float* b_proj = (const float*)d_in[4];
    const float* g1     = (const float*)d_in[5];
    const float* b1     = (const float*)d_in[6];
    const float* g2     = (const float*)d_in[7];
    const float* b2     = (const float*)d_in[8];
    const float* w_fc1  = (const float*)d_in[9];
    const float* b_fc1  = (const float*)d_in[10];
    const float* w_fc2  = (const float*)d_in[11];
    const float* b_fc2  = (const float*)d_in[12];

    char* ws = (char*)d_ws;
    unsigned short* xnbf   = (unsigned short*)(ws + o_xnbf);
    unsigned short* wqkvb  = (unsigned short*)(ws + o_wqkv);
    unsigned short* wprojb = (unsigned short*)(ws + o_wproj);
    unsigned short* wfc1b  = (unsigned short*)(ws + o_wfc1);
    unsigned short* wfc2b  = (unsigned short*)(ws + o_wfc2);
    float*          qkvf   = (float*)(ws + o_qkv);
    float*          attnf  = (float*)(ws + o_attn);
    unsigned short* attnsb = (unsigned short*)(ws + o_attns);
    float*          selx   = (float*)(ws + o_selx);
    float*          x2     = (float*)(ws + o_x2);
    float*          sraw   = (float*)(ws + o_sraw);
    int*            srcrow = (int*)(ws + o_src);
    float*          polb   = (float*)(ws + o_pol);
    // reused regions
    float*          xo     = attnf;                        // after gathers
    unsigned short* xobf   = xnbf;                         // after qkv gemm
    unsigned short* hbf    = attnsb;                       // after attn@v
    unsigned short* gelub  = (unsigned short*)(ws + o_qkv);// after attn@v

    // 1) weights -> bf16
    f32_to_bf16_kernel<<<2048, 256, 0, stream>>>(w_qkv,  wqkvb,  (size_t)Cc * C3);
    f32_to_bf16_kernel<<<1024, 256, 0, stream>>>(w_proj, wprojb, (size_t)Cc * Cc);
    f32_to_bf16_kernel<<<2048, 256, 0, stream>>>(w_fc1,  wfc1b,  (size_t)Cc * C4);
    f32_to_bf16_kernel<<<2048, 256, 0, stream>>>(w_fc2,  wfc2b,  (size_t)C4 * Cc);
    // 2) LN1 -> bf16
    layernorm_to_bf16<<<(int)BN, 256, 0, stream>>>(x, g1, b1, xnbf);
    // 3) qkv = ln(x) @ w_qkv, scaled by policy per row
    hipLaunchKernelGGL(HIP_KERNEL_NAME(gemm_bf16_wmma<0>), dim3(C3 / 16, 99), dim3(256), 0, stream,
                       xnbf, Cc, wqkvb, C3, nullptr, policy, nullptr,
                       qkvf, nullptr, (int)BN, Cc, C3);
    // 4) attn_raw = scale * q k^T
    attn_qk_wmma<<<dim3(13, 2, Bz * Hh), 256, 0, stream>>>(qkvf, attnf);
    // 5) softmax_with_policy (in place)
    softmax_policy_kernel<<<Bz * Hh * Nn, 256, 0, stream>>>(attnf, policy);
    // 6) token scores
    score_kernel<<<(int)BN, 256, 0, stream>>>(qkvf, attnf, sraw);
    // 7) ATS sampling: sort / cdf / inverse-transform / unique
    ats_sample_kernel<<<Bz, 256, 0, stream>>>(sraw, srcrow, polb);
    // 8) gathers
    gather_attn_kernel<<<dim3(Nn, Bz * Hh), 256, 0, stream>>>(attnf, srcrow, attnsb);
    gather_x_kernel<<<dim3(Nn, Bz), 256, 0, stream>>>(x, srcrow, selx);
    // 9) xo = attn_s @ v   (writes over attn region, already consumed)
    attn_av_wmma<<<dim3(HD / 16, 2, Bz * Hh), 256, 0, stream>>>(attnsb, qkvf, xo);
    // 10) xo -> bf16 (reuse LN1 buffer)
    f32_to_bf16_kernel<<<4096, 256, 0, stream>>>(xo, xobf, BN * Cc);
    // 11) x2 = (xo @ w_proj + b_proj) * pol + selected_x
    hipLaunchKernelGGL(HIP_KERNEL_NAME(gemm_bf16_wmma<1>), dim3(Cc / 16, 99), dim3(256), 0, stream,
                       xobf, Cc, wprojb, Cc, b_proj, polb, selx,
                       x2, nullptr, (int)BN, Cc, Cc);
    // 12) LN2 -> bf16 (reuse attn_s buffer)
    layernorm_to_bf16<<<(int)BN, 256, 0, stream>>>(x2, g2, b2, hbf);
    // 13) g = gelu(h @ w_fc1 + b_fc1) -> bf16 (reuse qkv buffer)
    hipLaunchKernelGGL(HIP_KERNEL_NAME(gemm_bf16_wmma<2>), dim3(C4 / 16, 99), dim3(256), 0, stream,
                       hbf, Cc, wfc1b, C4, b_fc1, nullptr, nullptr,
                       nullptr, gelub, (int)BN, Cc, C4);
    // 14) out = (x2 + g @ w_fc2 + b_fc2) * pol  -> d_out
    hipLaunchKernelGGL(HIP_KERNEL_NAME(gemm_bf16_wmma<3>), dim3(Cc / 16, 99), dim3(256), 0, stream,
                       gelub, C4, wfc2b, Cc, b_fc2, polb, x2,
                       (float*)d_out, nullptr, (int)BN, C4, Cc);
}
